// CompressedSwiGLUMLP_78786880078279
// MI455X (gfx1250) — compile-verified, matrix operand order checked by reading, not run on hardware
//
#include <hip/hip_runtime.h>

// ---------------------------------------------------------------------------
// SwiGLU MLP for MI455X (gfx1250): bf16 WMMA (v_wmma_f32_16x16x32_bf16),
// fp32 inputs converted on the fly, h_mid staged as bf16 in workspace.
// BK=64 (2 WMMA k-slices per staged tile), async global->LDS (ASYNCcnt)
// for the bf16 h_mid tile in kernel 2.
// ---------------------------------------------------------------------------

typedef __bf16 bf16_t;
typedef __attribute__((ext_vector_type(4)))  __bf16 v4bf;
typedef __attribute__((ext_vector_type(8)))  __bf16 v8bf;
typedef __attribute__((ext_vector_type(16))) __bf16 v16bf;
typedef __attribute__((ext_vector_type(8)))  float  v8f;
typedef __attribute__((ext_vector_type(4)))  float  v4f;
typedef __attribute__((ext_vector_type(4)))  int    v4i;

#define MDIM 4096   // B*S = 2*2048
#define HDIM 4096
#define IDIM 11008

#define BM 128      // tile rows (tokens)
#define BN 64       // tile cols (output features)
#define BK 64       // K-step = 2 WMMA k-slices of 32

#if defined(__HIP_DEVICE_COMPILE__) && __has_builtin(__builtin_amdgcn_global_load_async_to_lds_b128)
#define HAVE_ASYNC_LDS 1
#else
#define HAVE_ASYNC_LDS 0
#endif

#if HAVE_ASYNC_LDS
typedef __attribute__((address_space(1))) v4i* gas_v4i_ptr;
typedef __attribute__((address_space(3))) v4i* lds_v4i_ptr;

__device__ __forceinline__ void async_copy_b128(const void* g, void* l) {
    // prototype (from hipcc diagnostic): (v4i addrspace(1)*, v4i addrspace(3)*, Ii, Ii)
    gas_v4i_ptr gp = (gas_v4i_ptr)const_cast<void*>(g);
    lds_v4i_ptr lp = (lds_v4i_ptr)l;
    __builtin_amdgcn_global_load_async_to_lds_b128(gp, lp, /*offset=*/0, /*cpol=*/0);
}
__device__ __forceinline__ void wait_async0() {
#if __has_builtin(__builtin_amdgcn_s_wait_asynccnt)
    __builtin_amdgcn_s_wait_asynccnt(0);
#else
    asm volatile("s_wait_asynccnt 0" ::: "memory");
#endif
}
#endif

// ---------------------------------------------------------------------------
// Load a 16x32 bf16 WMMA fragment from an LDS tile stored [row][BK] row-major.
// CDNA5 16-bit A-matrix layout (mirrored for n-major B tiles):
//   lane L: row = L%16, half = L/16
//   element e (0..15): k = (e<8 ? e : e+8) + half*8
// -> two contiguous 16-byte ds_load_b128 runs per lane.
// kk = 0 or 32 selects the k-slice inside the BK=64 tile.
// ---------------------------------------------------------------------------
__device__ __forceinline__ v16bf load_frag(const bf16_t* lds, int rowBase, int kk, int lane) {
    const int r    = lane & 15;
    const int half = lane >> 4;
    const bf16_t* p = lds + (rowBase + r) * BK + kk + half * 8;
    v8bf lo = *(const v8bf*)(p);        // k = kk + half*8 .. +7
    v8bf hi = *(const v8bf*)(p + 16);   // k = kk + half*8+16 .. +23
    return __builtin_shufflevector(lo, hi, 0,1,2,3,4,5,6,7,8,9,10,11,12,13,14,15);
}

__device__ __forceinline__ v8f wmma_bf16(v16bf a, v16bf b, v8f c) {
    return __builtin_amdgcn_wmma_f32_16x16x32_bf16(
        /*neg_a=*/false, a, /*neg_b=*/false, b,
        /*c_mod=*/(short)0, c, /*reuse_a=*/false, /*reuse_b=*/false);
}

// ---------------------------------------------------------------------------
// Kernel 1: h_mid[m, n] = silu(x . Wg^T) * (x . Wu^T), n over IDIM
//   x  : [MDIM][HDIM] fp32       Wg, Wu : [IDIM][HDIM] fp32
//   out: h_mid [MDIM][IDIM] bf16 (workspace)
// Grid: (IDIM/BN, MDIM/BM), block = 256 threads (8 wave32 waves).
// ---------------------------------------------------------------------------
__global__ __launch_bounds__(256)
void swiglu_gate_up_kernel(const float* __restrict__ x,
                           const float* __restrict__ Wg,
                           const float* __restrict__ Wu,
                           bf16_t* __restrict__ hmid) {
    __shared__ __align__(16) bf16_t As[BM * BK];   // x tile (bf16)       16 KB
    __shared__ __align__(16) bf16_t Bg[BN * BK];   // Wg tile, n-major     8 KB
    __shared__ __align__(16) bf16_t Bu[BN * BK];   // Wu tile, n-major     8 KB

    const int tid   = threadIdx.x;
    const int lane  = tid & 31;
    const int wave  = tid >> 5;
    const int waveM = (wave & 3) * 32;   // 4 waves along M
    const int waveN = (wave >> 2) * 32;  // 2 waves along N

    const int blockN = blockIdx.x * BN;  // over IDIM
    const int blockM = blockIdx.y * BM;  // over MDIM

    v8f accG[2][2], accU[2][2];
#pragma unroll
    for (int r = 0; r < 2; ++r)
#pragma unroll
        for (int c = 0; c < 2; ++c) { accG[r][c] = (v8f)0.0f; accU[r][c] = (v8f)0.0f; }

    for (int k0 = 0; k0 < HDIM; k0 += BK) {
        // ---- stage x tile: 128x64 fp32 -> bf16 (8 float4 loads / thread) ----
#pragma unroll
        for (int i = 0; i < 8; ++i) {
            int e   = tid + i * 256;          // 2048 float4s
            int row = e >> 4;                 // 16 float4 per row
            int cv  = (e & 15) << 2;
            v4f v = *(const v4f*)(x + (size_t)(blockM + row) * HDIM + k0 + cv);
            v4bf b; b[0] = (bf16_t)v.x; b[1] = (bf16_t)v.y;
                    b[2] = (bf16_t)v.z; b[3] = (bf16_t)v.w;
            *(v4bf*)&As[row * BK + cv] = b;
        }
        // ---- stage Wg/Wu tiles: 64x64 each (4 float4 loads / thread each) ----
#pragma unroll
        for (int i = 0; i < 4; ++i) {
            int e   = tid + i * 256;          // 1024 float4s
            int row = e >> 4;
            int cv  = (e & 15) << 2;
            size_t off = (size_t)(blockN + row) * HDIM + k0 + cv;
            v4f g = *(const v4f*)(Wg + off);
            v4f u = *(const v4f*)(Wu + off);
            v4bf bg; bg[0] = (bf16_t)g.x; bg[1] = (bf16_t)g.y;
                     bg[2] = (bf16_t)g.z; bg[3] = (bf16_t)g.w;
            v4bf bu; bu[0] = (bf16_t)u.x; bu[1] = (bf16_t)u.y;
                     bu[2] = (bf16_t)u.z; bu[3] = (bf16_t)u.w;
            *(v4bf*)&Bg[row * BK + cv] = bg;
            *(v4bf*)&Bu[row * BK + cv] = bu;
        }
        // ---- prefetch next K tiles into cache (global_prefetch_b8) ----
        if (k0 + BK < HDIM) {
            __builtin_prefetch(x  + (size_t)(blockM + (tid & 127)) * HDIM + (k0 + BK), 0, 0);
            __builtin_prefetch(Wg + (size_t)(blockN + (tid & 63))  * HDIM + (k0 + BK), 0, 0);
            __builtin_prefetch(Wu + (size_t)(blockN + (tid & 63))  * HDIM + (k0 + BK), 0, 0);
        }
        __syncthreads();

        // ---- 2 k-slices x 8 WMMAs (2 rows x 2 cols x {gate,up}) ----
#pragma unroll
        for (int kk = 0; kk < BK; kk += 32) {
            v16bf a0  = load_frag(As, waveM,      kk, lane);
            v16bf a1  = load_frag(As, waveM + 16, kk, lane);
            v16bf bg0 = load_frag(Bg, waveN,      kk, lane);
            v16bf bg1 = load_frag(Bg, waveN + 16, kk, lane);
            v16bf bu0 = load_frag(Bu, waveN,      kk, lane);
            v16bf bu1 = load_frag(Bu, waveN + 16, kk, lane);

            accG[0][0] = wmma_bf16(a0, bg0, accG[0][0]);
            accG[0][1] = wmma_bf16(a0, bg1, accG[0][1]);
            accG[1][0] = wmma_bf16(a1, bg0, accG[1][0]);
            accG[1][1] = wmma_bf16(a1, bg1, accG[1][1]);
            accU[0][0] = wmma_bf16(a0, bu0, accU[0][0]);
            accU[0][1] = wmma_bf16(a0, bu1, accU[0][1]);
            accU[1][0] = wmma_bf16(a1, bu0, accU[1][0]);
            accU[1][1] = wmma_bf16(a1, bu1, accU[1][1]);
        }
        __syncthreads();
    }

    // ---- epilogue: silu(g)*u -> bf16, write h_mid ----
    // C/D layout: VGPR v, lane L -> m = v + 8*(L/16), n = L%16
    const int nL = lane & 15;
    const int mo = (lane >> 4) * 8;
#pragma unroll
    for (int r = 0; r < 2; ++r)
#pragma unroll
        for (int c = 0; c < 2; ++c) {
            v8f g = accG[r][c], u = accU[r][c];
            int m0 = blockM + waveM + r * 16 + mo;
            int n  = blockN + waveN + c * 16 + nL;
#pragma unroll
            for (int v = 0; v < 8; ++v) {
                float gv = g[v];
                float s  = gv / (1.0f + __expf(-gv));   // silu
                hmid[(size_t)(m0 + v) * IDIM + n] = (bf16_t)(s * u[v]);
            }
        }
}

// ---------------------------------------------------------------------------
// Kernel 2: y[m, n] = h_mid . Wd^T  (y[m,h] = sum_i hmid[m,i]*Wd[h,i])
//   hmid: [MDIM][IDIM] bf16     Wd: [HDIM][IDIM] fp32     y: [MDIM][HDIM] fp32
// The bf16 A tile goes straight to LDS via GLOBAL_LOAD_ASYNC_TO_LDS_B128
// (ASYNCcnt) when available — no VGPR round-trip.
// Grid: (HDIM/BN, MDIM/BM), block = 256 threads.
// ---------------------------------------------------------------------------
__global__ __launch_bounds__(256)
void swiglu_down_kernel(const bf16_t* __restrict__ hmid,
                        const float* __restrict__ Wd,
                        float* __restrict__ y) {
    __shared__ __align__(16) bf16_t As[BM * BK];   // h_mid tile, 16 KB
    __shared__ __align__(16) bf16_t Bs[BN * BK];   // Wd tile,     8 KB

    const int tid   = threadIdx.x;
    const int lane  = tid & 31;
    const int wave  = tid >> 5;
    const int waveM = (wave & 3) * 32;
    const int waveN = (wave >> 2) * 32;

    const int blockN = blockIdx.x * BN;  // over HDIM
    const int blockM = blockIdx.y * BM;  // over MDIM

    v8f acc[2][2];
#pragma unroll
    for (int r = 0; r < 2; ++r)
#pragma unroll
        for (int c = 0; c < 2; ++c) acc[r][c] = (v8f)0.0f;

    for (int k0 = 0; k0 < IDIM; k0 += BK) {
        // ---- stage hmid tile: 128x64 bf16 (1024 x 16B async copies) ----
#pragma unroll
        for (int i = 0; i < 4; ++i) {
            int e   = tid + i * 256;
            int row = e >> 3;                 // 8 x v8bf per row
            int cv  = (e & 7) << 3;
            const bf16_t* src = hmid + (size_t)(blockM + row) * IDIM + k0 + cv;
            bf16_t*       dst = &As[row * BK + cv];
#if HAVE_ASYNC_LDS
            async_copy_b128(src, dst);        // global -> LDS, ASYNCcnt
#else
            *(v8bf*)dst = *(const v8bf*)src;
#endif
        }
        // ---- stage Wd tile: 64x64 fp32 -> bf16 ----
#pragma unroll
        for (int i = 0; i < 4; ++i) {
            int e   = tid + i * 256;
            int row = e >> 4;
            int cv  = (e & 15) << 2;
            v4f w = *(const v4f*)(Wd + (size_t)(blockN + row) * IDIM + k0 + cv);
            v4bf b; b[0] = (bf16_t)w.x; b[1] = (bf16_t)w.y;
                    b[2] = (bf16_t)w.z; b[3] = (bf16_t)w.w;
            *(v4bf*)&Bs[row * BK + cv] = b;
        }
        if (k0 + BK < IDIM) {
            __builtin_prefetch(hmid + (size_t)(blockM + (tid & 127)) * IDIM + (k0 + BK), 0, 0);
            __builtin_prefetch(Wd   + (size_t)(blockN + (tid & 63))  * IDIM + (k0 + BK), 0, 0);
        }
#if HAVE_ASYNC_LDS
        wait_async0();                        // s_wait_asynccnt 0
#endif
        __syncthreads();

#pragma unroll
        for (int kk = 0; kk < BK; kk += 32) {
            v16bf a0 = load_frag(As, waveM,      kk, lane);
            v16bf a1 = load_frag(As, waveM + 16, kk, lane);
            v16bf b0 = load_frag(Bs, waveN,      kk, lane);
            v16bf b1 = load_frag(Bs, waveN + 16, kk, lane);

            acc[0][0] = wmma_bf16(a0, b0, acc[0][0]);
            acc[0][1] = wmma_bf16(a0, b1, acc[0][1]);
            acc[1][0] = wmma_bf16(a1, b0, acc[1][0]);
            acc[1][1] = wmma_bf16(a1, b1, acc[1][1]);
        }
        __syncthreads();
    }

    const int nL = lane & 15;
    const int mo = (lane >> 4) * 8;
#pragma unroll
    for (int r = 0; r < 2; ++r)
#pragma unroll
        for (int c = 0; c < 2; ++c) {
            int m0 = blockM + waveM + r * 16 + mo;
            int n  = blockN + waveN + c * 16 + nL;
#pragma unroll
            for (int v = 0; v < 8; ++v)
                y[(size_t)(m0 + v) * HDIM + n] = acc[r][c][v];
        }
}

// ---------------------------------------------------------------------------
extern "C" void kernel_launch(void* const* d_in, const int* in_sizes, int n_in,
                              void* d_out, int out_size, void* d_ws, size_t ws_size,
                              hipStream_t stream) {
    (void)in_sizes; (void)n_in; (void)out_size; (void)ws_size;

    const float* x  = (const float*)d_in[0];   // [2,2048,4096]
    const float* Wg = (const float*)d_in[1];   // [11008,4096]
    const float* Wu = (const float*)d_in[2];   // [11008,4096]
    const float* Wd = (const float*)d_in[3];   // [4096,11008]
    float*  y    = (float*)d_out;              // [2,2048,4096]
    bf16_t* hmid = (bf16_t*)d_ws;              // [4096,11008] bf16 (~86 MB)

    dim3 block(256, 1, 1);
    dim3 grid1(IDIM / BN, MDIM / BM, 1);       // 172 x 32
    swiglu_gate_up_kernel<<<grid1, block, 0, stream>>>(x, Wg, Wu, hmid);

    dim3 grid2(HDIM / BN, MDIM / BM, 1);       // 64 x 32
    swiglu_down_kernel<<<grid2, block, 0, stream>>>(hmid, Wd, y);
}
